// MotionGenerator_RNN_1563368096128
// MI455X (gfx1250) — compile-verified, matrix operand order checked by reading, not run on hardware
//
#include <hip/hip_runtime.h>
#include <cstdint>

// ---------------------------------------------------------------------------
// Types for CDNA5 WMMA (wave32): v_wmma_f32_16x16x32_bf16
// ---------------------------------------------------------------------------
typedef __attribute__((ext_vector_type(16))) __bf16        v16bf;
typedef __attribute__((ext_vector_type(8)))  float         v8f;
typedef __attribute__((ext_vector_type(8)))  unsigned int  v8u;

#if defined(__has_builtin)
# if __has_builtin(__builtin_amdgcn_global_load_async_to_lds_b128)
#  define HAS_ASYNC_LDS 1
# endif
#endif
#ifndef HAS_ASYNC_LDS
# define HAS_ASYNC_LDS 0
#endif

__device__ __forceinline__ unsigned int pack_bf16(float lo, float hi) {
  unsigned int bl = __builtin_bit_cast(unsigned int, lo);
  unsigned int bh = __builtin_bit_cast(unsigned int, hi);
  bl = (bl + 0x7FFFu + ((bl >> 16) & 1u)) >> 16;   // round-to-nearest-even
  bh = (bh + 0x7FFFu + ((bh >> 16) & 1u)) >> 16;
  return (bh << 16) | (bl & 0xFFFFu);
}

__device__ __forceinline__ float gelu_f(float x) {
  return 0.5f * x * (1.0f + erff(x * 0.70710678118654752f));
}

// Operand layout helpers (16-bit A/B WMMA layout, ISA 7.12.2):
//   lane<16 holds K {0..7,16..23}, lane>=16 holds K {8..15,24..31};
//   VGPR v<4: K=2v(+1) in its half, v>=4: K=16+2(v-4)(+1).
//   Packed tile = 32 lanes x 8 u32 = 256 u32 (1 KB), even-K in low half.
__device__ __forceinline__ int kk_of_v(int khalf, int v) {
  return khalf + ((v < 4) ? 2 * v : 16 + 2 * (v - 4));
}

// ---------------------------------------------------------------------------
// Weight pre-pack: f32 [N,K] row-major -> packed B tiles [nt][kt][lane][8]
// ---------------------------------------------------------------------------
__global__ __launch_bounds__(256) void pack_w_k(const float* __restrict__ W,
                                                unsigned int* __restrict__ dst,
                                                int N, int K) {
  int idx = blockIdx.x * 256 + threadIdx.x;
  int total = (N * K) >> 1;
  if (idx >= total) return;
  int v    = idx & 7;
  int lane = (idx >> 3) & 31;
  int tile = idx >> 8;
  int Kt   = K >> 5;
  int kt   = tile % Kt;
  int nt   = tile / Kt;
  int n    = nt * 16 + (lane & 15);
  int kk   = kt * 32 + kk_of_v((lane >> 4) * 8, v);
  const float* row = W + (size_t)n * K;
  dst[idx] = pack_bf16(row[kk], row[kk + 1]);
}

// ---------------------------------------------------------------------------
// A-matrix loader views (fused im2col / concat, f32 sources)
// ---------------------------------------------------------------------------
struct PlainA {                 // row-major [M, ld]
  const float* p; int ld;
  __device__ float operator()(int r, int k) const { return p[(size_t)r * ld + k]; }
};
struct Conv0A {                 // aud [32,128,600]; row=(b*32+n)*84+p, k=ci*7+t
  const float* aud;
  __device__ float operator()(int r, int k) const {
    int w = r / 84, p = r % 84, b = w >> 5, n = w & 31;
    int ci = k / 7, t = k % 7;
    return aud[((size_t)n * 128 + ci) * 600 + b * 30 + p + t];
  }
};
struct Conv1A {                 // x0 [(576*84),256]; k=ci*5+t
  const float* x0;
  __device__ float operator()(int r, int k) const {
    int w = r / 80, p = r % 80, ci = k / 5, t = k % 5;
    return x0[((size_t)w * 84 + p + t) * 256 + ci];
  }
};
struct Conv2A {                 // x1 [(576*80),256]; stride2 pad1, k=ci*4+t
  const float* x1;
  __device__ float operator()(int r, int k) const {
    int w = r / 40, p = r % 40, ci = k >> 2, t = k & 3;
    int pos = 2 * p - 1 + t;
    return (pos < 0 || pos >= 80) ? 0.0f : x1[((size_t)w * 80 + pos) * 256 + ci];
  }
};
struct CatCI {                  // [mo[:,:,29] (256) | lxm[:,:,0] (128)]
  const float* mo; const float* lxm;
  __device__ float operator()(int n, int k) const {
    return (k < 256) ? mo[((size_t)n * 256 + k) * 600 + 29]
                     : lxm[((size_t)n * 128 + (k - 256)) * 20];
  }
};
struct CatX {                   // [pre_mo(256) | a_t(256) | l_t(128)]
  const float* pre; const float* aud_seq; const float* lxm; int t;
  __device__ float operator()(int n, int k) const {
    if (k < 256)  return pre[n * 256 + k];
    if (k < 512)  return aud_seq[((size_t)t * 32 + n) * 256 + (k - 256)];
    return lxm[((size_t)n * 128 + (k - 512)) * 20 + 1 + t / 30];
  }
};

// ---------------------------------------------------------------------------
// A pre-pack: any A view -> packed A tiles [mt][ktStride ktOff+kt][lane][8]
// ---------------------------------------------------------------------------
template <class AL>
__global__ __launch_bounds__(256) void packA_k(AL al, unsigned int* __restrict__ dst,
                                               int Kt, int ktStride, int ktOff,
                                               int total) {
  int idx = blockIdx.x * 256 + threadIdx.x;
  if (idx >= total) return;
  int v    = idx & 7;
  int lane = (idx >> 3) & 31;
  int tile = idx >> 8;
  int kt   = tile % Kt;
  int mt   = tile / Kt;
  int m    = mt * 16 + (lane & 15);
  int kk   = kt * 32 + kk_of_v((lane >> 4) * 8, v);
  dst[((size_t)(mt * ktStride + ktOff + kt)) * 256 + (lane << 3) + v] =
      pack_bf16(al(m, kk), al(m, kk + 1));
}

// ---------------------------------------------------------------------------
// WMMA GEMM, both operands pre-packed bf16 (direct global-load version).
// Block: 256 thr = 8 wave32; block tile 32(M) x 64(N); wave tile 16x16.
// FUSE: 0 = bias, 1 = bias+gelu, 2 = bias + extra strided scatter to out2.
// ---------------------------------------------------------------------------
template <int FUSE>
__global__ __launch_bounds__(256) void gemm_pp_k(
    const unsigned int* __restrict__ apack, int aKtStride, int aKtOff,
    const unsigned int* __restrict__ packB, const float* __restrict__ bias,
    float* __restrict__ C, int ldc, int Kt,
    float* __restrict__ out2, int t) {
  const int lane  = threadIdx.x & 31;
  const int wid   = threadIdx.x >> 5;
  const int wm    = wid & 1;         // 2 M-tiles
  const int wn    = wid >> 1;        // 4 N-tiles
  const int Mbase = blockIdx.x * 32 + wm * 16;
  const int Nbase = blockIdx.y * 64 + wn * 16;
  const int mt    = blockIdx.x * 2 + wm;
  const int ntile = blockIdx.y * 4 + wn;

  const unsigned int* ap = apack + ((size_t)mt * aKtStride + aKtOff) * 256 + lane * 8;
  const unsigned int* bp = packB + ((size_t)ntile * Kt) * 256 + lane * 8;

  v8f acc = {};
  for (int kt = 0; kt < Kt; kt += 2) {
    __builtin_prefetch((const void*)(bp + 1024), 0, 1);   // 4 tiles ahead
    v8u a0 = *(const v8u*)ap;
    v8u b0 = *(const v8u*)bp;
    v8u a1 = *(const v8u*)(ap + 256);
    v8u b1 = *(const v8u*)(bp + 256);
    ap += 512; bp += 512;
    acc = __builtin_amdgcn_wmma_f32_16x16x32_bf16(
        false, __builtin_bit_cast(v16bf, a0), false, __builtin_bit_cast(v16bf, b0),
        (short)0, acc, false, false);
    acc = __builtin_amdgcn_wmma_f32_16x16x32_bf16(
        false, __builtin_bit_cast(v16bf, a1), false, __builtin_bit_cast(v16bf, b1),
        (short)0, acc, false, false);
  }

  const int col    = Nbase + (lane & 15);
  const int rowoff = (lane >> 4) * 8;      // lanes 16-31 hold M=8..15
  const float bv   = bias[col];
#pragma unroll
  for (int i = 0; i < 8; ++i) {
    float x = acc[i] + bv;
    if (FUSE == 1) x = gelu_f(x);
    int row = Mbase + rowoff + i;
    C[(size_t)row * ldc + col] = x;
    if (FUSE == 2) out2[((size_t)row * 256 + col) * 540 + t] = x;
  }
}

// ---------------------------------------------------------------------------
// Dual WMMA GEMM: two independent GEMMs in one launch (blockIdx.y selects).
// B operand staged through LDS with CDNA5 async loads
// (GLOBAL_LOAD_ASYNC_TO_LDS_B128 + s_wait_asynccnt, branch-free depth-4
//  pipeline: constant wait 6 in steady state, unrolled 6/4/2/0 drain).
// ---------------------------------------------------------------------------
struct GemmDesc {
  const unsigned int* ap; int aKtStride; int aKtOff;
  const unsigned int* bp; const float* bias; float* C; int ldc; int Kt;
};

#if HAS_ASYNC_LDS
typedef int v4i __attribute__((vector_size(16)));
typedef __attribute__((address_space(1))) v4i* gv4;
typedef __attribute__((address_space(3))) v4i* lv4;

__device__ __forceinline__ void async_issue_tile(const unsigned int* gtile,
                                                 unsigned int* ltile, int lane) {
  gv4 g = (gv4)(unsigned long long)(gtile + lane * 4);   // 16B per lane
  lv4 l = (lv4)(unsigned)(unsigned long long)(ltile + lane * 4);
  // IOFFSET is added to BOTH global and LDS addresses (ISA 10, async ops).
  __builtin_amdgcn_global_load_async_to_lds_b128(g, l, 0, 0);
  __builtin_amdgcn_global_load_async_to_lds_b128(g, l, 512, 0);
}
#endif

__global__ __launch_bounds__(256) void gemm_dual_k(GemmDesc d0, GemmDesc d1, int ny0) {
  const int lane  = threadIdx.x & 31;
  const int wid   = threadIdx.x >> 5;
  const bool firstHalf = ((int)blockIdx.y < ny0);
  GemmDesc d = firstHalf ? d0 : d1;
  const int by = firstHalf ? (int)blockIdx.y : (int)blockIdx.y - ny0;
  const int wm    = wid & 1;
  const int wn    = wid >> 1;
  const int Mbase = blockIdx.x * 32 + wm * 16;
  const int Nbase = by * 64 + wn * 16;
  const int mt    = blockIdx.x * 2 + wm;
  const int ntile = by * 4 + wn;

  const unsigned int* ap = d.ap + ((size_t)mt * d.aKtStride + d.aKtOff) * 256 + lane * 8;
  const unsigned int* bt = d.bp + ((size_t)ntile * d.Kt) * 256;   // B tile base

  v8f acc = {};
#if HAS_ASYNC_LDS
  __shared__ unsigned int lbuf[8][4][256];   // per-wave 4-slot pipeline, 32 KB
  unsigned int* slots = &lbuf[wid][0][0];
  // prologue: fill the 4-deep pipeline (dual GEMMs always have Kt >= 4)
#pragma unroll
  for (int p = 0; p < 4; ++p)
    async_issue_tile(bt + (size_t)p * 256, slots + p * 256, lane);
  int kt = 0;
  // steady state: constant wait, consume slot kt&3, refill same slot w/ kt+4
  for (; kt < d.Kt - 4; ++kt) {
    asm volatile("s_wait_asynccnt 0x6" ::: "memory");   // tile kt has landed
    v8u a0 = *(const v8u*)ap;  ap += 256;
    v8u b0 = *(const v8u*)(slots + (kt & 3) * 256 + lane * 8);  // ds_load_b128 x2
    acc = __builtin_amdgcn_wmma_f32_16x16x32_bf16(
        false, __builtin_bit_cast(v16bf, a0), false, __builtin_bit_cast(v16bf, b0),
        (short)0, acc, false, false);
    async_issue_tile(bt + (size_t)(kt + 4) * 256, slots + (kt & 3) * 256, lane);
  }
  // drain: 4 unrolled steps, compile-time-constant shrinking waits 6/4/2/0
#pragma unroll
  for (int e = 0; e < 4; ++e, ++kt) {
    if (e == 0)      asm volatile("s_wait_asynccnt 0x6" ::: "memory");
    else if (e == 1) asm volatile("s_wait_asynccnt 0x4" ::: "memory");
    else if (e == 2) asm volatile("s_wait_asynccnt 0x2" ::: "memory");
    else             asm volatile("s_wait_asynccnt 0x0" ::: "memory");
    v8u a0 = *(const v8u*)ap;  ap += 256;
    v8u b0 = *(const v8u*)(slots + (kt & 3) * 256 + lane * 8);
    acc = __builtin_amdgcn_wmma_f32_16x16x32_bf16(
        false, __builtin_bit_cast(v16bf, a0), false, __builtin_bit_cast(v16bf, b0),
        (short)0, acc, false, false);
  }
#else
  const unsigned int* bp = bt + lane * 8;
  for (int kt = 0; kt < d.Kt; kt += 2) {
    __builtin_prefetch((const void*)(bp + 1024), 0, 1);
    v8u a0 = *(const v8u*)ap;
    v8u b0 = *(const v8u*)bp;
    v8u a1 = *(const v8u*)(ap + 256);
    v8u b1 = *(const v8u*)(bp + 256);
    ap += 512; bp += 512;
    acc = __builtin_amdgcn_wmma_f32_16x16x32_bf16(
        false, __builtin_bit_cast(v16bf, a0), false, __builtin_bit_cast(v16bf, b0),
        (short)0, acc, false, false);
    acc = __builtin_amdgcn_wmma_f32_16x16x32_bf16(
        false, __builtin_bit_cast(v16bf, a1), false, __builtin_bit_cast(v16bf, b1),
        (short)0, acc, false, false);
  }
#endif

  const int col    = Nbase + (lane & 15);
  const int rowoff = (lane >> 4) * 8;
  const float bv   = d.bias[col];
#pragma unroll
  for (int i = 0; i < 8; ++i) {
    int row = Mbase + rowoff + i;
    d.C[(size_t)row * d.ldc + col] = acc[i] + bv;
  }
}

// ---------------------------------------------------------------------------
// Encoder layernorm + reshuffle into aud_seq[(b*30+f)*32+n][256]
// ---------------------------------------------------------------------------
__global__ __launch_bounds__(256) void enc_ln_k(const float* __restrict__ x3,
    const float* __restrict__ g, const float* __restrict__ bb,
    float* __restrict__ aud_seq) {
  int row  = blockIdx.x * 8 + (threadIdx.x >> 5);   // over 18*32*30 rows
  int lane = threadIdx.x & 31;
  int b = row / (32 * 30); int rem = row % (32 * 30);
  int n = rem / 30; int f = rem % 30;
  const float* src = x3 + ((size_t)((b * 32 + n) * 40 + f)) * 256;
  float vals[8]; float s = 0.f, s2 = 0.f;
#pragma unroll
  for (int j = 0; j < 8; ++j) {
    float x = src[lane * 8 + j]; vals[j] = x; s += x; s2 += x * x;
  }
#pragma unroll
  for (int m = 16; m > 0; m >>= 1) {
    s += __shfl_xor(s, m, 32); s2 += __shfl_xor(s2, m, 32);
  }
  float mu = s * (1.f / 256.f);
  float rs = rsqrtf(s2 * (1.f / 256.f) - mu * mu + 1e-5f);
  float* dst = aud_seq + ((size_t)((b * 30 + f) * 32 + n)) * 256;
#pragma unroll
  for (int j = 0; j < 8; ++j) {
    int c = lane * 8 + j;
    dst[c] = (vals[j] - mu) * rs * g[c] + bb[c];
  }
}

// ---------------------------------------------------------------------------
// Decoder layernorm of h1 -> DIRECTLY packed-A bf16 (for prediction GEMM).
// ---------------------------------------------------------------------------
__global__ __launch_bounds__(256) void dec_ln_pack_k(const float* __restrict__ hbuf,
    const float* __restrict__ g, const float* __restrict__ bb,
    unsigned int* __restrict__ apack) {
  int row  = blockIdx.x * 8 + (threadIdx.x >> 5);   // 0..31
  int lane = threadIdx.x & 31;
  const float* src = hbuf + (size_t)row * 2048 + 1024 + lane * 32;
  float vals[32]; float s = 0.f, s2 = 0.f;
#pragma unroll
  for (int j = 0; j < 32; ++j) {
    float x = src[j]; vals[j] = x; s += x; s2 += x * x;
  }
#pragma unroll
  for (int m = 16; m > 0; m >>= 1) {
    s += __shfl_xor(s, m, 32); s2 += __shfl_xor(s2, m, 32);
  }
  float mu = s * (1.f / 1024.f);
  float rs = rsqrtf(s2 * (1.f / 1024.f) - mu * mu + 1e-5f);
  int mt = row >> 4, lrow = row & 15;
  unsigned int* tilep = apack + ((size_t)(mt * 32 + lane)) * 256;  // kt == lane
#pragma unroll
  for (int p = 0; p < 16; ++p) {
    int kk = 2 * p;
    int c  = lane * 32 + kk;
    float v0 = (vals[kk]     - mu) * rs * g[c]     + bb[c];
    float v1 = (vals[kk + 1] - mu) * rs * g[c + 1] + bb[c + 1];
    int dl = lrow + ((kk >> 3) & 1) * 16;
    int dv = ((kk & 7) >> 1) + ((kk & 16) ? 4 : 0);
    tilep[dl * 8 + dv] = pack_bf16(v0, v1);
  }
}

// ---------------------------------------------------------------------------
// GRU gate combine, 2 elems/thread: updates f32 state AND emits packed-A bf16.
// ---------------------------------------------------------------------------
__global__ __launch_bounds__(256) void gru_comb_pack_k(const float* __restrict__ gi,
    const float* __restrict__ gh, float* __restrict__ hbuf, int hoff,
    unsigned int* __restrict__ apack) {
  int idx = blockIdx.x * 256 + threadIdx.x;   // 32 * 512
  int n = idx >> 9, j0 = (idx & 511) * 2;
  const float* a = gi + (size_t)n * 3072;
  const float* b = gh + (size_t)n * 3072;
  float* hp = hbuf + (size_t)n * 2048 + hoff;
  float hv[2];
#pragma unroll
  for (int e = 0; e < 2; ++e) {
    int j = j0 + e;
    float r  = 1.f / (1.f + expf(-(a[j] + b[j])));
    float z  = 1.f / (1.f + expf(-(a[1024 + j] + b[1024 + j])));
    float nn = tanhf(a[2048 + j] + r * b[2048 + j]);
    hv[e] = (1.f - z) * nn + z * hp[j];
    hp[j] = hv[e];
  }
  int kt = j0 >> 5, kk = j0 & 31;
  int dl = (n & 15) + ((kk >> 3) & 1) * 16;
  int dv = ((kk & 7) >> 1) + ((kk & 16) ? 4 : 0);
  int mt = n >> 4;
  apack[((size_t)(mt * 32 + kt)) * 256 + dl * 8 + dv] = pack_bf16(hv[0], hv[1]);
}

__global__ __launch_bounds__(256) void init_premo_k(const float* __restrict__ mo,
                                                    float* __restrict__ p0) {
  int idx = blockIdx.x * 256 + threadIdx.x;   // 32*256
  p0[idx] = mo[(size_t)idx * 600 + 29];
}

// ---------------------------------------------------------------------------
// Host-side launch helpers
// ---------------------------------------------------------------------------
typedef unsigned int u32;

template <int FUSE>
static inline void launch_gemm(const u32* ap, int aKtStride, int aKtOff,
                               const u32* pB, const float* bias, float* C, int ldc,
                               int M, int N, int K, hipStream_t s,
                               float* out2 = nullptr, int t = 0) {
  dim3 grid(M / 32, N / 64);
  gemm_pp_k<FUSE><<<grid, 256, 0, s>>>(ap, aKtStride, aKtOff, pB, bias, C, ldc,
                                       K >> 5, out2, t);
}

template <class AL>
static inline void launch_packA(AL al, u32* dst, int M, int K, int ktStride,
                                int ktOff, hipStream_t s) {
  int Kt = K >> 5;
  int total = (M / 16) * Kt * 256;
  packA_k<AL><<<(total + 255) / 256, 256, 0, s>>>(al, dst, Kt, ktStride, ktOff, total);
}

static inline void launch_pack(const float* W, u32* dst, int N, int K, hipStream_t s) {
  int total = (N * K) >> 1;
  pack_w_k<<<(total + 255) / 256, 256, 0, s>>>(W, dst, N, K);
}

extern "C" void kernel_launch(void* const* d_in, const int* in_sizes, int n_in,
                              void* d_out, int out_size, void* d_ws, size_t ws_size,
                              hipStream_t stream) {
  (void)in_sizes; (void)n_in; (void)out_size; (void)ws_size;

  const float* aud      = (const float*)d_in[0];
  const float* mo       = (const float*)d_in[1];
  const float* lxm      = (const float*)d_in[2];
  const float* ae_w0    = (const float*)d_in[3];
  const float* ae_b0    = (const float*)d_in[4];
  const float* ae_w1    = (const float*)d_in[5];
  const float* ae_b1    = (const float*)d_in[6];
  const float* ae_w2    = (const float*)d_in[7];
  const float* ae_b2    = (const float*)d_in[8];
  const float* ae_w3    = (const float*)d_in[9];
  const float* ae_b3    = (const float*)d_in[10];
  const float* ae_ln_g  = (const float*)d_in[11];
  const float* ae_ln_b  = (const float*)d_in[12];
  const float* ci_w0    = (const float*)d_in[13];
  const float* ci_b0    = (const float*)d_in[14];
  const float* ci_w1    = (const float*)d_in[15];
  const float* ci_b1    = (const float*)d_in[16];
  const float* ci_w2    = (const float*)d_in[17];
  const float* ci_b2    = (const float*)d_in[18];
  const float* dec_in_w = (const float*)d_in[19];
  const float* dec_in_b = (const float*)d_in[20];
  const float* gru_wih0 = (const float*)d_in[21];
  const float* gru_whh0 = (const float*)d_in[22];
  const float* gru_bih0 = (const float*)d_in[23];
  const float* gru_bhh0 = (const float*)d_in[24];
  const float* gru_wih1 = (const float*)d_in[25];
  const float* gru_whh1 = (const float*)d_in[26];
  const float* gru_bih1 = (const float*)d_in[27];
  const float* gru_bhh1 = (const float*)d_in[28];
  const float* dec_ln_g = (const float*)d_in[29];
  const float* dec_ln_b = (const float*)d_in[30];
  const float* pred_w   = (const float*)d_in[31];
  const float* pred_b   = (const float*)d_in[32];
  float* out = (float*)d_out;

  // ---- workspace carving -------------------------------------------------
  char* ws = (char*)d_ws;
  size_t off = 0;
  auto allocB = [&](size_t bytes) -> void* {
    void* p = ws + off;
    off += (bytes + 255) & ~(size_t)255;
    return p;
  };
  auto allocU = [&](size_t n) { return (u32*)allocB(n * 4); };
  auto allocF = [&](size_t n) { return (float*)allocB(n * 4); };

  // packed bf16 weights (u32 count = N*K/2)
  u32* p_conv0 = allocU(256u * 896  / 2);
  u32* p_conv1 = allocU(256u * 1280 / 2);
  u32* p_conv2 = allocU(256u * 1024 / 2);
  u32* p_w3    = allocU(256u * 256  / 2);
  u32* p_ci0   = allocU(1024u * 384  / 2);
  u32* p_ci1   = allocU(1024u * 1024 / 2);
  u32* p_ci2   = allocU(2048u * 1024 / 2);
  u32* p_dec   = allocU(1024u * 640  / 2);
  u32* p_wih0  = allocU(3072u * 1664 / 2);
  u32* p_whh0  = allocU(3072u * 1024 / 2);
  u32* p_wih1  = allocU(3072u * 1024 / 2);
  u32* p_whh1  = allocU(3072u * 1024 / 2);
  u32* p_pred  = allocU(256u * 1024 / 2);

  // activations (f32)
  float* x0      = allocF((size_t)48384 * 256);
  float* x1      = allocF((size_t)46080 * 256);
  float* x2      = allocF((size_t)23040 * 256);
  float* x3      = allocF((size_t)23040 * 256);
  float* aud_seq = allocF((size_t)540 * 32 * 256);
  float* emb     = allocF(32 * 1024);
  float* gi      = allocF(32 * 3072);
  float* gh      = allocF(32 * 3072);
  float* hbuf    = allocF(32 * 2048);             // [h0 | h1] per row
  float* ci1     = allocF(32 * 1024);
  float* ci2     = allocF(32 * 1024);
  float* pbufA   = allocF(32 * 256);
  float* pbufB   = allocF(32 * 256);
  float* pbuf[2] = { pbufA, pbufB };

  // packed-A scratch (u32)
  u32* apEnc = allocU((size_t)46080 * 1280 / 2);  // largest encoder A (reused)
  u32* apGin = allocU((size_t)2 * 52 * 256);      // [emb(kt 0..31) | xcat(kt 32..51)]
  u32* apH0  = allocU((size_t)2 * 32 * 256);      // packed h0 state
  u32* apH1  = allocU((size_t)2 * 32 * 256);      // packed h1 state
  u32* apLn  = allocU((size_t)2 * 32 * 256);      // packed layernorm(h1)
  u32* apD   = allocU((size_t)2 * 32 * 256);      // small generic A scratch

  // ---- 1. pack all weights to WMMA bf16 layout ---------------------------
  launch_pack(ae_w0,    p_conv0, 256, 896,  stream);
  launch_pack(ae_w1,    p_conv1, 256, 1280, stream);
  launch_pack(ae_w2,    p_conv2, 256, 1024, stream);
  launch_pack(ae_w3,    p_w3,    256, 256,  stream);
  launch_pack(ci_w0,    p_ci0,   1024, 384,  stream);
  launch_pack(ci_w1,    p_ci1,   1024, 1024, stream);
  launch_pack(ci_w2,    p_ci2,   2048, 1024, stream);
  launch_pack(dec_in_w, p_dec,   1024, 640,  stream);
  launch_pack(gru_wih0, p_wih0,  3072, 1664, stream);
  launch_pack(gru_whh0, p_whh0,  3072, 1024, stream);
  launch_pack(gru_wih1, p_wih1,  3072, 1024, stream);
  launch_pack(gru_whh1, p_whh1,  3072, 1024, stream);
  launch_pack(pred_w,   p_pred,  256, 1024, stream);

  // ---- 2. audio encoder: pack-A (fused im2col) then WMMA GEMM ------------
  launch_packA(Conv0A{aud}, apEnc, 48384, 896, 28, 0, stream);
  launch_gemm<1>(apEnc, 28, 0, p_conv0, ae_b0, x0, 256, 48384, 256, 896, stream);
  launch_packA(Conv1A{x0}, apEnc, 46080, 1280, 40, 0, stream);
  launch_gemm<1>(apEnc, 40, 0, p_conv1, ae_b1, x1, 256, 46080, 256, 1280, stream);
  launch_packA(Conv2A{x1}, apEnc, 23040, 1024, 32, 0, stream);
  launch_gemm<1>(apEnc, 32, 0, p_conv2, ae_b2, x2, 256, 23040, 256, 1024, stream);
  launch_packA(PlainA{x2, 256}, apEnc, 23040, 256, 8, 0, stream);
  launch_gemm<1>(apEnc, 8, 0, p_w3, ae_b3, x3, 256, 23040, 256, 256, stream);
  enc_ln_k<<<2160, 256, 0, stream>>>(x3, ae_ln_g, ae_ln_b, aud_seq);

  // ---- 3. cell-state init ------------------------------------------------
  launch_packA(CatCI{mo, lxm}, apD, 32, 384, 12, 0, stream);
  launch_gemm<1>(apD, 12, 0, p_ci0, ci_b0, ci1, 1024, 32, 1024, 384, stream);
  launch_packA(PlainA{ci1, 1024}, apD, 32, 1024, 32, 0, stream);
  launch_gemm<1>(apD, 32, 0, p_ci1, ci_b1, ci2, 1024, 32, 1024, 1024, stream);
  launch_packA(PlainA{ci2, 1024}, apD, 32, 1024, 32, 0, stream);
  launch_gemm<0>(apD, 32, 0, p_ci2, ci_b2, hbuf, 2048, 32, 2048, 1024, stream);
  launch_packA(PlainA{hbuf, 2048},        apH0, 32, 1024, 32, 0, stream);
  launch_packA(PlainA{hbuf + 1024, 2048}, apH1, 32, 1024, 32, 0, stream);
  init_premo_k<<<32, 256, 0, stream>>>(mo, pbuf[0]);

  // ---- 4. sequential decoder: 540 steps ----------------------------------
  for (int t = 0; t < 540; ++t) {
    const float* pre = pbuf[t & 1];
    float* pnext = pbuf[(t + 1) & 1];

    // xcat -> gin buffer k-tiles 32..51 (K=640)
    launch_packA(CatX{pre, aud_seq, lxm, t}, apGin, 32, 640, 52, 32, stream);
    // emb = gelu(xcat @ dec_in_w.T + b)  (reads xcat sub-range of apGin)
    launch_gemm<1>(apGin, 52, 32, p_dec, dec_in_b, emb, 1024, 32, 1024, 640, stream);
    // emb -> gin buffer k-tiles 0..31
    launch_packA(PlainA{emb, 1024}, apGin, 32, 1024, 52, 0, stream);

    // GRU layer 0: gi = gin@wih0, gh = h0@whh0 -- ONE dual launch
    GemmDesc dgi0 = { apGin, 52, 0, p_wih0, gru_bih0, gi, 3072, 52 };
    GemmDesc dgh0 = { apH0,  32, 0, p_whh0, gru_bhh0, gh, 3072, 32 };
    gemm_dual_k<<<dim3(1, 96), 256, 0, stream>>>(dgi0, dgh0, 48);
    gru_comb_pack_k<<<64, 256, 0, stream>>>(gi, gh, hbuf, 0, apH0);

    // GRU layer 1: gi = h0n@wih1, gh = h1@whh1 -- ONE dual launch
    GemmDesc dgi1 = { apH0, 32, 0, p_wih1, gru_bih1, gi, 3072, 32 };
    GemmDesc dgh1 = { apH1, 32, 0, p_whh1, gru_bhh1, gh, 3072, 32 };
    gemm_dual_k<<<dim3(1, 96), 256, 0, stream>>>(dgi1, dgh1, 48);
    gru_comb_pack_k<<<64, 256, 0, stream>>>(gi, gh, hbuf, 1024, apH1);

    // prediction head: layernorm(h1) -> packed A, then GEMM w/ fused scatter
    dec_ln_pack_k<<<4, 256, 0, stream>>>(hbuf, dec_ln_g, dec_ln_b, apLn);
    launch_gemm<2>(apLn, 32, 0, p_pred, pred_b, pnext, 256, 32, 256, 1024, stream,
                   out, t);
  }
}
